// TemporalRiskGNN_23106924053184
// MI455X (gfx1250) — matrix operand order repaired and checked by reading, not compiled, not run on hardware
//
#include <hip/hip_runtime.h>
#include <hip/hip_bf16.h>
#include <math.h>

typedef __attribute__((ext_vector_type(16))) __bf16 bf16x16;
typedef __attribute__((ext_vector_type(8)))  float  f32x8;

#define N_NODES 50000
#define T_STEPS 4
#define F_DIM   128
#define E_EDGES 800000
#define NF      (N_NODES * F_DIM)

#define MODE_SIGMOID 0
#define MODE_HR      1   // out = H * sigmoid(acc+bias)
#define MODE_GRU     2   // out = Z*H + (1-Z)*tanh(acc+bias)
#define MODE_RELU    3
#define MODE_NONE    4

// ---------------------------------------------------------------------------
// Weight swizzle: f32 (K x 128) -> bf16 wave32 WMMA B-fragment layout.
// out[((kg*8 + ct)*32 + lane)*16 + j] = W[kg*32 + (j>>3)*16 + (lane>>4)*8 + (j&7)][ct*16 + (lane&15)]
// k <  ksplit -> Wa[k*128+col]; k >= ksplit -> Wb[(k-ksplit)*128+col]
// ---------------------------------------------------------------------------
__global__ void swizzle_w_kernel(const float* __restrict__ Wa,
                                 const float* __restrict__ Wb,
                                 __bf16* __restrict__ out,
                                 int ktot, int ksplit) {
  int tid = blockIdx.x * blockDim.x + threadIdx.x;
  if (tid >= ktot * 128) return;
  int j    = tid & 15;
  int lane = (tid >> 4) & 31;
  int ct   = (tid >> 9) & 7;
  int kg   = tid >> 12;
  int k    = kg * 32 + ((j >> 3) << 4) + ((lane >> 4) << 3) + (j & 7);
  int col  = ct * 16 + (lane & 15);
  float v = (k < ksplit) ? Wa[k * 128 + col] : Wb[(k - ksplit) * 128 + col];
  out[tid] = (__bf16)v;
}

// ---------------------------------------------------------------------------
// Sparse graph kernels
// ---------------------------------------------------------------------------
__global__ void degree_kernel(const long long* __restrict__ src,
                              float* __restrict__ deg, int ne) {
  int e = blockIdx.x * blockDim.x + threadIdx.x;
  if (e < ne) unsafeAtomicAdd(&deg[src[e]], 1.0f);
}

__global__ void edgew_kernel(const long long* __restrict__ src,
                             const long long* __restrict__ dst,
                             const float* __restrict__ deg,
                             float* __restrict__ w, int ne) {
  int e = blockIdx.x * blockDim.x + threadIdx.x;
  if (e >= ne) return;
  float ds = deg[src[e]];
  float dd = deg[dst[e]];
  float is = ds > 0.f ? rsqrtf(ds) : 0.f;
  float id = dd > 0.f ? rsqrtf(dd) : 0.f;
  w[e] = -is * id;
}

// One wave per edge: 32 lanes x float4 = 128 features.
__global__ void prop_kernel(const float* __restrict__ in, float* out,
                            const long long* __restrict__ src,
                            const long long* __restrict__ dst,
                            const float* __restrict__ w, int ne) {
  int wid  = (int)((blockIdx.x * (unsigned)blockDim.x + threadIdx.x) >> 5);
  int lane = threadIdx.x & 31;
  if (wid >= ne) return;
  long long s = src[wid];
  long long d = dst[wid];
  float wt = w[wid];
  float4 v = *((const float4*)(in + (size_t)s * F_DIM) + lane);
  float* po = out + (size_t)d * F_DIM + lane * 4;
  unsafeAtomicAdd(po + 0, wt * v.x);
  unsafeAtomicAdd(po + 1, wt * v.y);
  unsafeAtomicAdd(po + 2, wt * v.z);
  unsafeAtomicAdd(po + 3, wt * v.w);
}

// Tx2 = 2*prop(Tx1) - Tx0
__global__ void cheb_finish_kernel(float* __restrict__ t2,
                                   const float* __restrict__ t0, int n) {
  int i = blockIdx.x * blockDim.x + threadIdx.x;
  if (i < n) t2[i] = 2.0f * t2[i] - t0[i];
}

// ---------------------------------------------------------------------------
// Fused basis-GEMM + activation epilogue, NB/MODE compile-time.
// out(N x 128) = act( sum_b basis[b](N x 128) @ W[b](128 x 128) + bias )
// 256 threads = 8 waves; block owns 64 rows x 128 cols.
// Double-buffered LDS staging (9KB x2), one barrier per 32-deep K chunk.
// ---------------------------------------------------------------------------
template <int NB, int MODE>
__launch_bounds__(256)
__global__ void wmma_gemm_kernel(
    const float* __restrict__ b0, const float* __restrict__ b1,
    const float* __restrict__ b2, const float* __restrict__ b3,
    const float* __restrict__ b4, const float* __restrict__ b5,
    const __bf16* __restrict__ Wsw,
    const float* __restrict__ bias0, const float* __restrict__ bias1,
    const float* zbuf, const float* hbuf,
    float* out, int nrows) {
  __shared__ float lds[2][64 * 36];
  const float* const basis[6] = {b0, b1, b2, b3, b4, b5};
  constexpr int NCH = NB * 4;        // 32-deep K chunks

  int tid     = threadIdx.x;
  int wv      = tid >> 5;
  int lane    = tid & 31;
  int rowBase = blockIdx.x * 64;
  int m16     = (wv & 3) * 16;
  int colBase = (wv >> 2) * 64;

  f32x8 acc[4] = {};

  int  srow  = tid >> 2;             // staging: 4 threads/row, 8 floats each
  int  scol  = (tid & 3) * 8;
  int  grow  = rowBase + srow;
  bool rowok = grow < nrows;
  int  mloc  = m16 + (lane & 15);
  int  khalf = lane >> 4;

  // prologue: stage chunk 0 into buffer 0
  {
    float4 v0 = make_float4(0.f, 0.f, 0.f, 0.f), v1 = v0;
    if (rowok) {
      const float* p = basis[0] + (size_t)grow * F_DIM + scol;
      v0 = *(const float4*)p;
      v1 = *(const float4*)(p + 4);
    }
    *(float4*)&lds[0][srow * 36 + scol]     = v0;
    *(float4*)&lds[0][srow * 36 + scol + 4] = v1;
  }
  __syncthreads();

  #pragma unroll
  for (int c = 0; c < NCH; ++c) {
    const int buf = c & 1;

    // issue next chunk's global loads early (overlap with WMMA)
    float4 n0 = make_float4(0.f, 0.f, 0.f, 0.f), n1 = n0;
    if (c + 1 < NCH) {
      if (rowok) {
        const float* p =
            basis[(c + 1) >> 2] + (size_t)grow * F_DIM + ((c + 1) & 3) * 32 + scol;
        n0 = *(const float4*)p;
        n1 = *(const float4*)(p + 4);
      }
    }

    // A fragment: a[j]=A[m][khalf*8+j], a[8+j]=A[m][16+khalf*8+j]
    bf16x16 afrag;
    {
      const float* lp = &lds[buf][mloc * 36 + khalf * 8];
      #pragma unroll
      for (int j = 0; j < 8; ++j) {
        afrag[j]     = (__bf16)lp[j];
        afrag[8 + j] = (__bf16)lp[16 + j];
      }
    }

    #pragma unroll
    for (int ct = 0; ct < 4; ++ct) {
      int cglob = (colBase >> 4) + ct;
      const bf16x16 bfrag =
          *(const bf16x16*)(Wsw + (((size_t)(c * 8 + cglob) * 32 + lane) << 4));
      acc[ct] = __builtin_amdgcn_wmma_f32_16x16x32_bf16(
          false, afrag, false, bfrag, (short)0, acc[ct], false, false);
    }

    if (c + 1 < NCH) {
      *(float4*)&lds[buf ^ 1][srow * 36 + scol]     = n0;
      *(float4*)&lds[buf ^ 1][srow * 36 + scol + 4] = n1;
      __syncthreads();
    }
  }

  // Epilogue: C/D layout -> element (m = khalf*8+r, n = lane&15) per tile.
  #pragma unroll
  for (int ct = 0; ct < 4; ++ct) {
    int col = colBase + ct * 16 + (lane & 15);
    float bsum = bias0 ? bias0[col] : 0.0f;
    if (bias1) bsum += bias1[col];
    #pragma unroll
    for (int r = 0; r < 8; ++r) {
      int row = rowBase + m16 + khalf * 8 + r;
      if (row < nrows) {
        float val  = acc[ct][r] + bsum;
        size_t idx = (size_t)row * F_DIM + col;
        if constexpr (MODE == MODE_SIGMOID) {
          out[idx] = 1.0f / (1.0f + __expf(-val));
        } else if constexpr (MODE == MODE_HR) {
          out[idx] = hbuf[idx] * (1.0f / (1.0f + __expf(-val)));
        } else if constexpr (MODE == MODE_GRU) {
          float ht = tanhf(val);
          float z  = zbuf[idx];
          out[idx] = z * hbuf[idx] + (1.0f - z) * ht;
        } else if constexpr (MODE == MODE_RELU) {
          out[idx] = fmaxf(val, 0.0f);
        } else {
          out[idx] = val;
        }
      }
    }
  }
}

// ---------------------------------------------------------------------------
extern "C" void kernel_launch(void* const* d_in, const int* in_sizes, int n_in,
                              void* d_out, int out_size, void* d_ws, size_t ws_size,
                              hipStream_t stream) {
  const float*     x   = (const float*)d_in[0];
  const long long* ei  = (const long long*)d_in[1];
  const float* Wxz = (const float*)d_in[2];
  const float* Whz = (const float*)d_in[3];
  const float* Wxr = (const float*)d_in[4];
  const float* Whr = (const float*)d_in[5];
  const float* Wxh = (const float*)d_in[6];
  const float* Whh = (const float*)d_in[7];
  const float* bxz = (const float*)d_in[8];
  const float* bhz = (const float*)d_in[9];
  const float* bxr = (const float*)d_in[10];
  const float* bhr = (const float*)d_in[11];
  const float* bxh = (const float*)d_in[12];
  const float* bhh = (const float*)d_in[13];
  const float* Wd1 = (const float*)d_in[14];
  const float* bd1 = (const float*)d_in[15];
  const float* Wd2 = (const float*)d_in[16];
  const float* bd2 = (const float*)d_in[17];
  float* outp = (float*)d_out;

  float* ws = (float*)d_ws;
  float* H   = ws;
  float* X1  = ws + (size_t)1 * NF;
  float* X2  = ws + (size_t)2 * NF;
  float* B1  = ws + (size_t)3 * NF;   // H-basis 1, reused as (H*R)-basis 1
  float* B2  = ws + (size_t)4 * NF;
  float* Z   = ws + (size_t)5 * NF;
  float* HR  = ws + (size_t)6 * NF;
  float* deg = ws + (size_t)7 * NF;
  float* wE  = deg + N_NODES;
  __bf16* Wz_sw  = (__bf16*)(wE + E_EDGES);
  __bf16* Wr_sw  = Wz_sw + 768 * 128;
  __bf16* Wh_sw  = Wr_sw + 768 * 128;
  __bf16* Wd1_sw = Wh_sw + 768 * 128;
  __bf16* Wd2_sw = Wd1_sw + 128 * 128;

  swizzle_w_kernel<<<384, 256, 0, stream>>>(Wxz, Whz, Wz_sw, 768, 384);
  swizzle_w_kernel<<<384, 256, 0, stream>>>(Wxr, Whr, Wr_sw, 768, 384);
  swizzle_w_kernel<<<384, 256, 0, stream>>>(Wxh, Whh, Wh_sw, 768, 384);
  swizzle_w_kernel<<<64,  256, 0, stream>>>(Wd1, Wd1, Wd1_sw, 128, 128);
  swizzle_w_kernel<<<64,  256, 0, stream>>>(Wd2, Wd2, Wd2_sw, 128, 128);

  hipMemsetAsync(H, 0, (size_t)NF * sizeof(float), stream);

  const int EB = (E_EDGES + 255) / 256;
  const int PB = E_EDGES / 8;              // 8 edge-waves per 256-thread block
  const int FB = (NF + 255) / 256;
  const int GB = (N_NODES + 63) / 64;

  for (int t = 0; t < T_STEPS; ++t) {
    const float*     xt  = x + (size_t)t * NF;
    const long long* src = ei + (size_t)t * 2 * E_EDGES;
    const long long* dst = src + E_EDGES;

    hipMemsetAsync(deg, 0, N_NODES * sizeof(float), stream);
    degree_kernel<<<EB, 256, 0, stream>>>(src, deg, E_EDGES);
    edgew_kernel<<<EB, 256, 0, stream>>>(src, dst, deg, wE, E_EDGES);

    // x Chebyshev basis (shared by Z, R, Ht)
    hipMemsetAsync(X1, 0, (size_t)NF * 4, stream);
    prop_kernel<<<PB, 256, 0, stream>>>(xt, X1, src, dst, wE, E_EDGES);
    hipMemsetAsync(X2, 0, (size_t)NF * 4, stream);
    prop_kernel<<<PB, 256, 0, stream>>>(X1, X2, src, dst, wE, E_EDGES);
    cheb_finish_kernel<<<FB, 256, 0, stream>>>(X2, xt, NF);

    // H Chebyshev basis (shared by Z, R)
    hipMemsetAsync(B1, 0, (size_t)NF * 4, stream);
    prop_kernel<<<PB, 256, 0, stream>>>(H, B1, src, dst, wE, E_EDGES);
    hipMemsetAsync(B2, 0, (size_t)NF * 4, stream);
    prop_kernel<<<PB, 256, 0, stream>>>(B1, B2, src, dst, wE, E_EDGES);
    cheb_finish_kernel<<<FB, 256, 0, stream>>>(B2, H, NF);

    // Z = sigmoid(...); HR = H * sigmoid(R-preact)
    wmma_gemm_kernel<6, MODE_SIGMOID><<<GB, 256, 0, stream>>>(
        xt, X1, X2, H, B1, B2, Wz_sw, bxz, bhz, nullptr, nullptr, Z, N_NODES);
    wmma_gemm_kernel<6, MODE_HR><<<GB, 256, 0, stream>>>(
        xt, X1, X2, H, B1, B2, Wr_sw, bxr, bhr, nullptr, H, HR, N_NODES);

    // (H*R) Chebyshev basis (reuse B1/B2)
    hipMemsetAsync(B1, 0, (size_t)NF * 4, stream);
    prop_kernel<<<PB, 256, 0, stream>>>(HR, B1, src, dst, wE, E_EDGES);
    hipMemsetAsync(B2, 0, (size_t)NF * 4, stream);
    prop_kernel<<<PB, 256, 0, stream>>>(B1, B2, src, dst, wE, E_EDGES);
    cheb_finish_kernel<<<FB, 256, 0, stream>>>(B2, HR, NF);

    // H = Z*H + (1-Z)*tanh(...)   (row-local in-place update, race-free)
    wmma_gemm_kernel<6, MODE_GRU><<<GB, 256, 0, stream>>>(
        xt, X1, X2, HR, B1, B2, Wh_sw, bxh, bhh, Z, H, H, N_NODES);
  }

  // decoder: X1 = relu(H@Wd1+bd1); out = X1@Wd2+bd2; then append H
  wmma_gemm_kernel<1, MODE_RELU><<<GB, 256, 0, stream>>>(
      H, H, H, H, H, H, Wd1_sw, bd1, nullptr, nullptr, nullptr, X1, N_NODES);
  wmma_gemm_kernel<1, MODE_NONE><<<GB, 256, 0, stream>>>(
      X1, X1, X1, X1, X1, X1, Wd2_sw, bd2, nullptr, nullptr, nullptr, outp, N_NODES);
  hipMemcpyAsync(outp + (size_t)NF, H, (size_t)NF * sizeof(float),
                 hipMemcpyDeviceToDevice, stream);
}
